// LSTMModel_1099511628183
// MI455X (gfx1250) — compile-verified, matrix operand order checked by reading, not run on hardware
//
#include <hip/hip_runtime.h>

#define B_    4096
#define S_    512
#define IN_   2
#define H_    64
#define TILES 2            // batch tiles (of 16 rows) per workgroup; 4 waves per tile

typedef __attribute__((ext_vector_type(16))) _Float16 v16h;
typedef __attribute__((ext_vector_type(8)))  _Float16 v8h;
typedef __attribute__((ext_vector_type(4)))  _Float16 v4h;
typedef __attribute__((ext_vector_type(8)))  float    v8f;
typedef __attribute__((ext_vector_type(4)))  float    v4f;

// sigmoid / tanh with v_exp_f32 + v_rcp_f32 (both TRANS ops, co-execute with WMMA)
static __device__ __forceinline__ float fsig(float x) {
    float e = __expf(-x);
    return __builtin_amdgcn_rcpf(1.0f + e);
}
static __device__ __forceinline__ float ftanh(float x) {
    float e = __expf(2.0f * x);
    return __builtin_fmaf(-2.0f, __builtin_amdgcn_rcpf(e + 1.0f), 1.0f);
}

static __device__ __forceinline__ v16h zero16h() {
    v16h r;
#pragma unroll
    for (int j = 0; j < 16; ++j) r[j] = (_Float16)0.0f;
    return r;
}

// Load a 16-bit WMMA fragment from a row-major f16 LDS matrix with row stride 64.
// Used both for A (H buffer, row = M) and B (weight buffer, row = N).
// ISA 16-bit fragment layout:
//   lanes 0-15 : row = lane,    halves 0..7 = K koff..+7, 8..15 = K koff+16..+23
//   lanes16-31 : row = lane-16, koff += 8
static __device__ __forceinline__ v16h lds_frag(const _Float16* M, int row_base,
                                                int kh, int lane) {
    const int row  = row_base + (lane & 15);
    const int koff = kh * 32 + ((lane & 16) ? 8 : 0);
    const _Float16* p = M + row * 64 + koff;
    v8h lo = *(const v8h*)(p);
    v8h hi = *(const v8h*)(p + 16);
    v16h r;
#pragma unroll
    for (int j = 0; j < 8; ++j) { r[j] = lo[j]; r[8 + j] = hi[j]; }
    return r;
}

// Load a B-fragment [32(K) x 16(N)] directly from a row-major f32 global weight
// matrix W[N_total][64], converting f32 -> f16 (used once, for resident frags).
static __device__ __forceinline__ v16h load_bfrag(const float* __restrict__ W,
                                                  int Nb, int kh, int lane) {
    const int N    = Nb + (lane & 15);
    const int koff = kh * 32 + ((lane & 16) ? 8 : 0);
    const float* base = W + (size_t)N * 64 + koff;
    v4f f0 = ((const v4f*)base)[0];
    v4f f1 = ((const v4f*)base)[1];
    v4f f2 = ((const v4f*)(base + 16))[0];
    v4f f3 = ((const v4f*)(base + 16))[1];
    v16h r;
#pragma unroll
    for (int j = 0; j < 4; ++j) {
        r[j]      = (_Float16)f0[j];
        r[4 + j]  = (_Float16)f1[j];
        r[8 + j]  = (_Float16)f2[j];
        r[12 + j] = (_Float16)f3[j];
    }
    return r;
}

#define WMMA_F16(A, Bf, C) \
    __builtin_amdgcn_wmma_f32_16x16x32_f16(false, (A), false, (Bf), (short)0, (C), false, false)

__global__ __launch_bounds__(32 * 4 * TILES)
void lstm2_fused_wmma(const float* __restrict__ x,
                      const float* __restrict__ Wih0, const float* __restrict__ Whh0,
                      const float* __restrict__ bih0, const float* __restrict__ bhh0,
                      const float* __restrict__ Wih1, const float* __restrict__ Whh1,
                      const float* __restrict__ bih1, const float* __restrict__ bhh1,
                      const float* __restrict__ fcw,  const float* __restrict__ fcb,
                      float* __restrict__ out) {
    // f16 copies of W_hh0 / W_ih1, row-major [256][64] (B-fragments loaded per step)
    __shared__ alignas(16) _Float16 W0l[256 * 64];
    __shared__ alignas(16) _Float16 W1l[256 * 64];
    __shared__ alignas(16) _Float16 H0s[TILES][16 * 64];   // layer0 hidden exchange
    __shared__ alignas(16) _Float16 H1s[TILES][16 * 64];   // layer1 hidden exchange
    __shared__ float2 xb[TILES][16];                       // x[t] broadcast

    const int lane    = threadIdx.x & 31;
    const int wid     = threadIdx.x >> 5;
    const int tile    = wid >> 2;                          // 0..TILES-1
    const int w       = wid & 3;                           // N-slice within tile
    const int rowbase = (blockIdx.x * TILES + tile) * 16;  // batch row base
    const int colq    = lane & 15;                         // column within 16-wide tile
    const int rbase   = (lane & 16) ? 8 : 0;               // row base for D layout
    const int hcol    = w * 16 + colq;                     // owned hidden column

    // ---- prologue: convert W_hh0, W_ih1 to f16 in LDS (cooperative) ----
    for (int i = threadIdx.x; i < 256 * 64 / 4; i += blockDim.x) {
        v4f a = ((const v4f*)Whh0)[i];
        v4f b = ((const v4f*)Wih1)[i];
        v4h ah, bh;
#pragma unroll
        for (int j = 0; j < 4; ++j) { ah[j] = (_Float16)a[j]; bh[j] = (_Float16)b[j]; }
        ((v4h*)W0l)[i] = ah;
        ((v4h*)W1l)[i] = bh;
    }

    // Per-lane scalars: combined biases and the tiny IN=2 input projection of layer 0.
    float bias0[4], bias1[4], wa[4], wb[4];
#pragma unroll
    for (int gt = 0; gt < 4; ++gt) {
        const int g = gt * 64 + w * 16 + colq;
        bias0[gt] = bih0[g] + bhh0[g];
        bias1[gt] = bih1[g] + bhh1[g];
        wa[gt]    = Wih0[g * 2 + 0];
        wb[gt]    = Wih0[g * 2 + 1];
    }

    // Layer-1 recurrent weights stay resident in VGPRs (8 frags = 64 VGPRs).
    v16h F2[4][2];
#pragma unroll
    for (int gt = 0; gt < 4; ++gt) {
        const int Nb = gt * 64 + w * 16;
        F2[gt][0] = load_bfrag(Whh1, Nb, 0, lane);
        F2[gt][1] = load_bfrag(Whh1, Nb, 1, lane);
    }

    // Recurrent state: h as A-fragments (f16), c in D-layout f32 registers.
    v16h a00 = zero16h(), a01 = zero16h();   // layer0 h (K=0..31, 32..63)
    v16h a10 = zero16h(), a11 = zero16h();   // layer1 h
    v8f c0, c1;
#pragma unroll
    for (int q = 0; q < 8; ++q) { c0[q] = 0.0f; c1[q] = 0.0f; }

    _Float16* Hp0 = &H0s[tile][0];
    _Float16* Hp1 = &H1s[tile][0];

    __syncthreads();   // LDS weights ready

    for (int t = 0; t < S_; ++t) {
        // broadcast x[rowbase..rowbase+15][t][0:2] through LDS
        if (w == 0 && lane < 16) {
            const float* xp = x + ((size_t)(rowbase + lane) * S_ + t) * IN_;
            xb[tile][lane] = make_float2(xp[0], xp[1]);
            if (t + 1 < S_) __builtin_prefetch(xp + IN_, 0, 0);
        }
        __syncthreads();

        // ---------------- layer 0 ----------------
        v8f acc[4];
#pragma unroll
        for (int gt = 0; gt < 4; ++gt) {
#pragma unroll
            for (int q = 0; q < 8; ++q) {
                float2 xv = xb[tile][rbase + q];
                acc[gt][q] = __builtin_fmaf(wb[gt], xv.y,
                             __builtin_fmaf(wa[gt], xv.x, bias0[gt]));
            }
        }
#pragma unroll
        for (int gt = 0; gt < 4; ++gt) {
            const int Nb = gt * 64 + w * 16;
            v16h b0 = lds_frag(W0l, Nb, 0, lane);
            v16h b1 = lds_frag(W0l, Nb, 1, lane);
            acc[gt] = WMMA_F16(a00, b0, acc[gt]);
            acc[gt] = WMMA_F16(a01, b1, acc[gt]);
        }
        // cell update: i/f/g/o are same-lane/same-slot across the 4 acc tiles
#pragma unroll
        for (int q = 0; q < 8; ++q) {
            float ig = fsig(acc[0][q]);
            float fg = fsig(acc[1][q]);
            float gg = ftanh(acc[2][q]);
            float og = fsig(acc[3][q]);
            float c  = __builtin_fmaf(fg, c0[q], ig * gg);
            c0[q] = c;
            Hp0[(rbase + q) * 64 + hcol] = (_Float16)(og * ftanh(c));
        }
        __syncthreads();
        a00 = lds_frag(Hp0, 0, 0, lane);
        a01 = lds_frag(Hp0, 0, 1, lane);

        // ---------------- layer 1 ----------------
#pragma unroll
        for (int gt = 0; gt < 4; ++gt) {
#pragma unroll
            for (int q = 0; q < 8; ++q) acc[gt][q] = bias1[gt];
        }
#pragma unroll
        for (int gt = 0; gt < 4; ++gt) {
            const int Nb = gt * 64 + w * 16;
            v16h b0 = lds_frag(W1l, Nb, 0, lane);
            v16h b1 = lds_frag(W1l, Nb, 1, lane);
            acc[gt] = WMMA_F16(a00, b0, acc[gt]);          // input = layer0 h
            acc[gt] = WMMA_F16(a01, b1, acc[gt]);
            acc[gt] = WMMA_F16(a10, F2[gt][0], acc[gt]);   // recurrent
            acc[gt] = WMMA_F16(a11, F2[gt][1], acc[gt]);
        }
#pragma unroll
        for (int q = 0; q < 8; ++q) {
            float ig = fsig(acc[0][q]);
            float fg = fsig(acc[1][q]);
            float gg = ftanh(acc[2][q]);
            float og = fsig(acc[3][q]);
            float c  = __builtin_fmaf(fg, c1[q], ig * gg);
            c1[q] = c;
            Hp1[(rbase + q) * 64 + hcol] = (_Float16)(og * ftanh(c));
        }
        __syncthreads();
        a10 = lds_frag(Hp1, 0, 0, lane);
        a11 = lds_frag(Hp1, 0, 1, lane);
    }

    // ---------------- final FC: out[b] = h_last . fc_w + fc_b ----------------
    if (w == 0 && lane < 16) {
        float s = fcb[0];
        const _Float16* hp = Hp1 + lane * 64;
#pragma unroll
        for (int c = 0; c < 64; ++c) s += (float)hp[c] * fcw[c];
        out[rowbase + lane] = s;
    }
}

extern "C" void kernel_launch(void* const* d_in, const int* in_sizes, int n_in,
                              void* d_out, int out_size, void* d_ws, size_t ws_size,
                              hipStream_t stream) {
    (void)in_sizes; (void)n_in; (void)out_size; (void)d_ws; (void)ws_size;
    const float* x    = (const float*)d_in[0];
    const float* Wih0 = (const float*)d_in[1];
    const float* Whh0 = (const float*)d_in[2];
    const float* bih0 = (const float*)d_in[3];
    const float* bhh0 = (const float*)d_in[4];
    const float* Wih1 = (const float*)d_in[5];
    const float* Whh1 = (const float*)d_in[6];
    const float* bih1 = (const float*)d_in[7];
    const float* bhh1 = (const float*)d_in[8];
    const float* fcw  = (const float*)d_in[9];
    const float* fcb  = (const float*)d_in[10];
    float* out = (float*)d_out;

    dim3 grid(B_ / (16 * TILES));      // 128 workgroups
    dim3 block(32 * 4 * TILES);        // 256 threads = 8 wave32s
    lstm2_fused_wmma<<<grid, block, 0, stream>>>(x, Wih0, Whh0, bih0, bhh0,
                                                 Wih1, Whh1, bih1, bhh1,
                                                 fcw, fcb, out);
}
